// So3kratesInteractionBlock_62053687492649
// MI455X (gfx1250) — matrix operand order repaired and checked by reading, not compiled, not run on hardware
//
#include <hip/hip_runtime.h>

typedef __attribute__((ext_vector_type(16))) __bf16 v16bf;
typedef __attribute__((ext_vector_type(8)))  float  v8f;
typedef __attribute__((ext_vector_type(4)))  float  v4f;

// Geometry: y = [x | d_chi] -> K = 131 (pad 160 = 5 chunks of 32)
//           out cols = 131   (pad 144 = 9 tiles of 16)
#define KDIM 131
#define JDIM 131
#define NKC  5
#define NJT  9
// Per-fragment LDS stride: 32 lanes * 32B + 4 * 16B pad = 1088 B (16B aligned)
#define FRAG_STRIDE 1088
// kc-major layout: base(kc) = kc * (NJT*2*FRAG_STRIDE); within a kc block,
// tile jt's hi frag at jt*2*FRAG_STRIDE, lo frag at +FRAG_STRIDE.
// Max DS immediate offset used: 8*2176 + 1104 = 18,512 B < 64 KB.
#define KC_STRIDE   (NJT * 2 * FRAG_STRIDE)          // 19,584 B
#define LDS_BYTES   (NKC * KC_STRIDE)                // 97,920 B

__device__ __forceinline__ void bf16_split(float v, __bf16& h, __bf16& l) {
    h = (__bf16)v;                 // RNE f32 -> bf16
    l = (__bf16)(v - (float)h);    // residual
}

__global__ __launch_bounds__(256) void so3k_gemm_kernel(
    const float* __restrict__ x,      // (N,128)
    const float* __restrict__ chi,    // (N,15)
    const float* __restrict__ W,      // (131,131) row-major
    float* __restrict__ a1,           // (N,128)
    float* __restrict__ bws,          // (N,3) scratch: mixed[:,128:131]
    int nTiles)                       // N/16
{
    __shared__ __align__(16) unsigned char smem[LDS_BYTES];

    // ---- Cooperative fill: W -> bf16 hi/lo B-fragments in LDS (kc-major) ----
    // B (32x16, bf16) layout: lanes 0-15 hold N=lane, K=0..7,16..23 (2/VGPR);
    // lanes 16-31 hold N=lane-16, K=8..15,24..31.  B[k][n] = W[n][k].
    for (int idx = threadIdx.x; idx < NJT * NKC * 512; idx += 256) {
        int e    = idx & 15;
        int lane = (idx >> 4) & 31;
        int fid  = idx >> 9;               // kc*NJT + jt
        int kc   = fid / NJT;
        int jt   = fid - kc * NJT;
        int n    = jt * 16 + (lane & 15);
        int k    = kc * 32 + (lane >> 4) * 8 + (e & 7) + (e >> 3) * 16;
        float w  = (n < JDIM && k < KDIM) ? W[n * KDIM + k] : 0.0f;
        __bf16 hi, lo;
        bf16_split(w, hi, lo);
        int off  = lane * 32 + (lane >> 3) * 16 + e * 2;
        int base = kc * KC_STRIDE + jt * (2 * FRAG_STRIDE);
        *(__bf16*)(smem + base + off) = hi;
        *(__bf16*)(smem + base + FRAG_STRIDE + off) = lo;
    }
    __syncthreads();

    const int lane    = threadIdx.x & 31;
    const int wv      = threadIdx.x >> 5;
    const int col     = lane & 15;         // N index within tile / row for A
    const int khalf   = lane >> 4;         // which K half this lane holds (A)
    const int laneOff = lane * 32 + (lane >> 3) * 16;

    for (int t = blockIdx.x * 8 + wv; t < nTiles; t += gridDim.x * 8) {
        const int r0  = t * 16;
        const int row = r0 + col;

        // per-row segment sums of chi^2 (segments 3,5,7)
        const float* cp = chi + (long long)row * 15;
        float d0 = 0.f, d1 = 0.f, d2 = 0.f;
        #pragma unroll
        for (int m = 0; m < 3;  ++m) { float c = cp[m]; d0 += c * c; }
        #pragma unroll
        for (int m = 3; m < 8;  ++m) { float c = cp[m]; d1 += c * c; }
        #pragma unroll
        for (int m = 8; m < 15; ++m) { float c = cp[m]; d2 += c * c; }

        v8f acc[NJT];
        #pragma unroll
        for (int j = 0; j < NJT; ++j)
            #pragma unroll
            for (int q = 0; q < 8; ++q) acc[j][q] = 0.0f;

        const float* xrow = x + (long long)row * 128;

        // Keep this a real loop: caps VGPR pressure < 256 (no vgpr-msb churn,
        // higher wave occupancy for an HBM-bound kernel). Each iteration still
        // exposes 27 WMMAs + 36 ds_load_b128 + 4 global_load_b128 of ILP.
        #pragma unroll 1
        for (int kc = 0; kc < NKC; ++kc) {
            // ---- Build A fragment (16x32 bf16) hi/lo for this K-chunk ----
            // lane: M = lane&15; elems 0..7 -> K = kc*32 + khalf*8 + e,
            //                    elems 8..15 -> +16 more.
            union { v16bf v; __bf16 e[16]; } ah, al;
            if (kc < 4) {
                const int kb = kc * 32 + khalf * 8;
                v4f q0 = *(const v4f*)(xrow + kb);
                v4f q1 = *(const v4f*)(xrow + kb + 4);
                v4f q2 = *(const v4f*)(xrow + kb + 16);
                v4f q3 = *(const v4f*)(xrow + kb + 20);
                #pragma unroll
                for (int i = 0; i < 4; ++i) {
                    bf16_split(q0[i], ah.e[i],      al.e[i]);
                    bf16_split(q1[i], ah.e[i + 4],  al.e[i + 4]);
                    bf16_split(q2[i], ah.e[i + 8],  al.e[i + 8]);
                    bf16_split(q3[i], ah.e[i + 12], al.e[i + 12]);
                }
            } else {
                // K = 128..159: only 128..130 valid (= d_chi), rest zero.
                #pragma unroll
                for (int i = 0; i < 16; ++i) { ah.e[i] = (__bf16)0.f; al.e[i] = (__bf16)0.f; }
                if (khalf == 0) {
                    bf16_split(d0, ah.e[0], al.e[0]);   // K=128
                    bf16_split(d1, ah.e[1], al.e[1]);   // K=129
                    bf16_split(d2, ah.e[2], al.e[2]);   // K=130
                }
            }

            // One runtime base per kc; all 9 tiles via DS immediate offsets.
            const unsigned char* fp = smem + kc * KC_STRIDE + laneOff;

            // ---- 9 output col-tiles: 3-product bf16 split accumulation ----
            #pragma unroll
            for (int jt = 0; jt < NJT; ++jt) {
                const int fb = jt * (2 * FRAG_STRIDE);
                union { v16bf v; uint4 q[2]; } bh, bl;
                bh.q[0] = *(const uint4*)(fp + fb);
                bh.q[1] = *(const uint4*)(fp + fb + 16);
                bl.q[0] = *(const uint4*)(fp + fb + FRAG_STRIDE);
                bl.q[1] = *(const uint4*)(fp + fb + FRAG_STRIDE + 16);
                acc[jt] = __builtin_amdgcn_wmma_f32_16x16x32_bf16(
                    false, ah.v, false, bh.v, (short)0, acc[jt], false, false);
                acc[jt] = __builtin_amdgcn_wmma_f32_16x16x32_bf16(
                    false, ah.v, false, bl.v, (short)0, acc[jt], false, false);
                acc[jt] = __builtin_amdgcn_wmma_f32_16x16x32_bf16(
                    false, al.v, false, bh.v, (short)0, acc[jt], false, false);
            }
        }

        // ---- Epilogue: C/D layout: lane -> N=col, VGPR v -> M = khalf*8+v ----
        const int mbase = khalf * 8;
        #pragma unroll
        for (int jt = 0; jt < 8; ++jt) {
            #pragma unroll
            for (int v = 0; v < 8; ++v) {
                a1[(long long)(r0 + mbase + v) * 128 + jt * 16 + col] = acc[jt][v];
            }
        }
        if (col < 3) {   // tile 8: only cols 128..130 are real -> scratch
            #pragma unroll
            for (int v = 0; v < 8; ++v)
                bws[(long long)(r0 + mbase + v) * 3 + col] = acc[8][v];
        }
    }
}

// b1_ = repeat(b1, (3,5,7)) * chi, fully coalesced over N*15 elements
__global__ __launch_bounds__(256) void so3k_expand_kernel(
    const float* __restrict__ bws,   // (N,3)
    const float* __restrict__ chi,   // (N,15)
    float* __restrict__ outb,        // (N,15)
    long long total)
{
    long long i = (long long)blockIdx.x * 256 + threadIdx.x;
    if (i >= total) return;
    int n = (int)(i / 15);
    int m = (int)(i - (long long)n * 15);
    int seg = (m < 3) ? 0 : ((m < 8) ? 1 : 2);
    outb[i] = bws[(long long)n * 3 + seg] * chi[i];
}

extern "C" void kernel_launch(void* const* d_in, const int* in_sizes, int n_in,
                              void* d_out, int out_size, void* d_ws, size_t ws_size,
                              hipStream_t stream)
{
    const float* x   = (const float*)d_in[0];   // (N,128)
    const float* chi = (const float*)d_in[1];   // (N,15)
    const float* W   = (const float*)d_in[2];   // (131,131)
    const int Nrows  = in_sizes[0] / 128;

    float* a1   = (float*)d_out;
    float* outb = (float*)d_out + (long long)Nrows * 128;
    float* bws  = (float*)d_ws;                 // needs Nrows*3*4 = 12 MB

    const int nTiles = Nrows / 16;              // N = 1e6 -> 62,500 exact
    int blocks = 2048;
    if (blocks * 8 > nTiles) blocks = (nTiles + 7) / 8;
    if (blocks < 1) blocks = 1;
    so3k_gemm_kernel<<<blocks, 256, 0, stream>>>(x, chi, W, a1, bws, nTiles);

    const long long total = (long long)Nrows * 15;
    const int b2 = (int)((total + 255) / 256);
    so3k_expand_kernel<<<b2, 256, 0, stream>>>(bws, chi, outb, total);
}
